// QuestionGuidedCompressor_41102837023209
// MI455X (gfx1250) — matrix operand order repaired and checked by reading, not compiled, not run on hardware
//
#include <hip/hip_runtime.h>
#include <hip/hip_bf16.h>

// ---------------------------------------------------------------------------
// QuestionGuidedCompressor for MI455X (gfx1250, wave32, WMMA)
// B=16, N=4096, D=768, K=2048
// 2M x 2N register-tiled WMMA: each A/B fragment feeds two v_wmma ops,
// halving both LDS (A) and L2 (B) traffic per FLOP vs the naive scheme.
// ---------------------------------------------------------------------------

#define BB     16
#define NN     4096
#define DD     768
#define KSEL   2048
#define LDS_STRIDE 776   // 768 + 8 halves pad: breaks 64-bank conflict on row loads

typedef __attribute__((ext_vector_type(16))) _Float16 v16h;
typedef __attribute__((ext_vector_type(8)))  _Float16 v8h;
typedef __attribute__((ext_vector_type(8)))  float    v8f;

__device__ __forceinline__ float gelu_exact(float x) {
    return 0.5f * x * (1.0f + erff(x * 0.70710678118654752f));
}
__device__ __forceinline__ float sigmoidf(float x) {
    return 1.0f / (1.0f + __expf(-x));
}

// A-fragment (16-bit A 16x32, contiguous k order shared with B): lane m=lane&15, half=lane>>4.
// VGPR0..3 <- K = half*16 + 0..7 ; VGPR4..7 <- K = half*16 + 8..15.
// => two contiguous 8-half chunks at (kbase + half*16) and (kbase + half*16 + 8).
__device__ __forceinline__ v16h load_afrag(const _Float16* tbase, int ks) {
    v8h c0 = *(const v8h*)(tbase + ks * 32);
    v8h c1 = *(const v8h*)(tbase + ks * 32 + 8);
    return __builtin_shufflevector(c0, c1, 0,1,2,3,4,5,6,7,8,9,10,11,12,13,14,15);
}

__device__ __forceinline__ v8f wmma16(v16h a, v16h b, v8f c) {
    return __builtin_amdgcn_wmma_f32_16x16x32_f16(false, a, false, b, (short)0, c,
                                                  false, false);
}

// ---------------------------------------------------------------------------
// Pack: out[j*768 + d] = (f16) in[d*outdim + j]   (transpose to [out][in] f16)
// ---------------------------------------------------------------------------
__global__ void pack_transpose_f16(const float* __restrict__ in,
                                   _Float16* __restrict__ out, int outdim) {
    int idx = blockIdx.x * blockDim.x + threadIdx.x;
    int total = outdim * DD;
    if (idx < total) {
        int j = idx / DD;
        int d = idx - j * DD;
        out[idx] = (_Float16)in[(size_t)d * outdim + j];
    }
}

// ---------------------------------------------------------------------------
// qrel[b][j] = b_rel1[j] + sum_d q[b][d] * W_rel1[(D+d)][j]
// ---------------------------------------------------------------------------
__global__ void qrel_kernel(const float* __restrict__ q,
                            const float* __restrict__ Wrel1,
                            const float* __restrict__ brel1,
                            float* __restrict__ qrel) {
    int idx = blockIdx.x * blockDim.x + threadIdx.x;
    if (idx >= BB * DD) return;
    int b = idx / DD, j = idx - b * DD;
    const float* qs = q + (size_t)b * DD;
    float acc = brel1[j];
    for (int d = 0; d < DD; ++d)
        acc += qs[d] * Wrel1[(size_t)(DD + d) * DD + j];
    qrel[idx] = acc;
}

// ---------------------------------------------------------------------------
// Score kernel: 1 wave per 32-token super-tile (two 16-row WMMA tiles).
//   relevance = sigmoid( gelu(V@W1a + qrel) @ w2 + b2 )
//   imp       = sigmoid( gelu(V@Wimp + bimp1) @ wimp2 + bimp2 )
//   sim       = cosine(V, q)
//   final     = max(relevance, 0.7*sim + 0.3*imp)
// ---------------------------------------------------------------------------
__global__ __launch_bounds__(32) void score_kernel(
    const float* __restrict__ V, const float* __restrict__ q,
    const _Float16* __restrict__ W1aT, const float* __restrict__ qrel,
    const float* __restrict__ w2, const float* __restrict__ b2,
    const _Float16* __restrict__ WimpT, const float* __restrict__ bimp1,
    const float* __restrict__ wimp2, const float* __restrict__ bimp2,
    float* __restrict__ fsout)
{
    __shared__ _Float16 tile[32 * LDS_STRIDE];
    __shared__ float sdot[32], snrm[32];

    const int t    = blockIdx.x;
    const int b    = t >> 7;            // N/32 = 128 super-tiles per batch
    const int m0   = (t & 127) * 32;
    const int lane = threadIdx.x;
    const int lm   = lane & 15;
    const int half = lane >> 4;

    // ---- preload q row, accumulate ||q||^2 --------------------------------
    float qreg[24];
    float qn_p = 0.f;
    #pragma unroll
    for (int j = 0; j < 24; ++j) {
        float v = q[(size_t)b * DD + lane + 32 * j];
        qreg[j] = v; qn_p += v * v;
    }
    for (int mask = 16; mask; mask >>= 1) qn_p += __shfl_xor(qn_p, mask, 32);
    const float qn = fmaxf(sqrtf(qn_p), 1e-8f);

    // ---- stage 32 rows (f32 -> f16 LDS) + sim statistics ------------------
    for (int m = 0; m < 32; ++m) {
        const float* src = V + ((size_t)b * NN + m0 + m) * DD;
        float da = 0.f, na = 0.f;
        #pragma unroll
        for (int j = 0; j < 24; ++j) {
            float v = src[lane + 32 * j];
            tile[m * LDS_STRIDE + lane + 32 * j] = (_Float16)v;
            da += v * qreg[j];
            na += v * v;
        }
        for (int mask = 16; mask; mask >>= 1) {
            da += __shfl_xor(da, mask, 32);
            na += __shfl_xor(na, mask, 32);
        }
        if (lane == 0) { sdot[m] = da; snrm[m] = na; }
    }
    __syncthreads();

    const _Float16* tbase0 = &tile[lm * LDS_STRIDE + half * 16];
    const _Float16* tbase1 = tbase0 + 16 * LDS_STRIDE;

    // ---- relevance: 48 n-tiles (2 at a time) x 24 k-steps -----------------
    float dr0[8], dr1[8];
    #pragma unroll
    for (int r = 0; r < 8; ++r) { dr0[r] = 0.f; dr1[r] = 0.f; }

    for (int nt2 = 0; nt2 < 24; ++nt2) {
        v8f a00 = {}, a01 = {}, a10 = {}, a11 = {};
        const _Float16* bp0 = W1aT + (size_t)(nt2 * 32 + lm) * DD + half * 16;
        const _Float16* bp1 = bp0 + (size_t)16 * DD;
        __builtin_prefetch(bp0 + (size_t)32 * DD, 0, 0);   // next n-tile pair
        #pragma unroll 2
        for (int ks = 0; ks < 24; ++ks) {
            v16h A0 = load_afrag(tbase0, ks);
            v16h A1 = load_afrag(tbase1, ks);
            v16h B0 = *(const v16h*)(bp0 + ks * 32);
            v16h B1 = *(const v16h*)(bp1 + ks * 32);
            a00 = wmma16(A0, B0, a00);
            a01 = wmma16(A0, B1, a01);
            a10 = wmma16(A1, B0, a10);
            a11 = wmma16(A1, B1, a11);
        }
        int n0 = nt2 * 32 + lm, n1 = n0 + 16;
        float q0 = qrel[(size_t)b * DD + n0];
        float q1 = qrel[(size_t)b * DD + n1];
        float w0 = w2[n0], w1 = w2[n1];
        #pragma unroll
        for (int r = 0; r < 8; ++r) {
            dr0[r] += gelu_exact(a00[r] + q0) * w0 + gelu_exact(a01[r] + q1) * w1;
            dr1[r] += gelu_exact(a10[r] + q0) * w0 + gelu_exact(a11[r] + q1) * w1;
        }
    }

    float rel0[8], rel1[8];
    {
        const float brel2 = b2[0];
        #pragma unroll
        for (int r = 0; r < 8; ++r) {
            float v0 = dr0[r], v1 = dr1[r];
            v0 += __shfl_xor(v0, 1, 32); v1 += __shfl_xor(v1, 1, 32);
            v0 += __shfl_xor(v0, 2, 32); v1 += __shfl_xor(v1, 2, 32);
            v0 += __shfl_xor(v0, 4, 32); v1 += __shfl_xor(v1, 4, 32);
            v0 += __shfl_xor(v0, 8, 32); v1 += __shfl_xor(v1, 8, 32);
            rel0[r] = sigmoidf(v0 + brel2);
            rel1[r] = sigmoidf(v1 + brel2);
        }
    }

    // ---- importance: 24 n-tiles (2 at a time) x 24 k-steps ----------------
    float di0[8], di1[8];
    #pragma unroll
    for (int r = 0; r < 8; ++r) { di0[r] = 0.f; di1[r] = 0.f; }

    for (int nt2 = 0; nt2 < 12; ++nt2) {
        v8f a00 = {}, a01 = {}, a10 = {}, a11 = {};
        const _Float16* bp0 = WimpT + (size_t)(nt2 * 32 + lm) * DD + half * 16;
        const _Float16* bp1 = bp0 + (size_t)16 * DD;
        __builtin_prefetch(bp0 + (size_t)32 * DD, 0, 0);
        #pragma unroll 2
        for (int ks = 0; ks < 24; ++ks) {
            v16h A0 = load_afrag(tbase0, ks);
            v16h A1 = load_afrag(tbase1, ks);
            v16h B0 = *(const v16h*)(bp0 + ks * 32);
            v16h B1 = *(const v16h*)(bp1 + ks * 32);
            a00 = wmma16(A0, B0, a00);
            a01 = wmma16(A0, B1, a01);
            a10 = wmma16(A1, B0, a10);
            a11 = wmma16(A1, B1, a11);
        }
        int n0 = nt2 * 32 + lm, n1 = n0 + 16;
        float c0 = bimp1[n0], c1 = bimp1[n1];
        float w0 = wimp2[n0], w1 = wimp2[n1];
        #pragma unroll
        for (int r = 0; r < 8; ++r) {
            di0[r] += gelu_exact(a00[r] + c0) * w0 + gelu_exact(a01[r] + c1) * w1;
            di1[r] += gelu_exact(a10[r] + c0) * w0 + gelu_exact(a11[r] + c1) * w1;
        }
    }

    float imp0[8], imp1[8];
    {
        const float bi2 = bimp2[0];
        #pragma unroll
        for (int r = 0; r < 8; ++r) {
            float v0 = di0[r], v1 = di1[r];
            v0 += __shfl_xor(v0, 1, 32); v1 += __shfl_xor(v1, 1, 32);
            v0 += __shfl_xor(v0, 2, 32); v1 += __shfl_xor(v1, 2, 32);
            v0 += __shfl_xor(v0, 4, 32); v1 += __shfl_xor(v1, 4, 32);
            v0 += __shfl_xor(v0, 8, 32); v1 += __shfl_xor(v1, 8, 32);
            imp0[r] = sigmoidf(v0 + bi2);
            imp1[r] = sigmoidf(v1 + bi2);
        }
    }

    // ---- combine + write final scores -------------------------------------
    if (lm == 0) {
        #pragma unroll
        for (int r = 0; r < 8; ++r) {
            int m = r + 8 * half;
            {
                float vn  = fmaxf(sqrtf(snrm[m]), 1e-8f);
                float sim = sdot[m] / (vn * qn);
                float comb = 0.7f * sim + 0.3f * imp0[r];
                fsout[(size_t)b * NN + m0 + m] = fmaxf(rel0[r], comb);
            }
            {
                float vn  = fmaxf(sqrtf(snrm[16 + m]), 1e-8f);
                float sim = sdot[16 + m] / (vn * qn);
                float comb = 0.7f * sim + 0.3f * imp1[r];
                fsout[(size_t)b * NN + m0 + 16 + m] = fmaxf(rel1[r], comb);
            }
        }
    }
}

// ---------------------------------------------------------------------------
// Top-k via full bitonic sort (score desc, tie: index asc == jax top_k order)
// ---------------------------------------------------------------------------
__global__ __launch_bounds__(512) void topk_kernel(
    const float* __restrict__ fs, float* __restrict__ oscore,
    float* __restrict__ oidxf, int* __restrict__ oidxi)
{
    __shared__ float s[NN];
    __shared__ int   si[NN];
    const int b = blockIdx.x, tid = threadIdx.x;

    for (int i = tid; i < NN; i += 512) { s[i] = fs[(size_t)b * NN + i]; si[i] = i; }
    __syncthreads();

    for (int size = 2; size <= NN; size <<= 1) {
        for (int stride = size >> 1; stride > 0; stride >>= 1) {
            for (int t2 = tid; t2 < NN / 2; t2 += 512) {
                int i = ((t2 / stride) * stride * 2) + (t2 % stride);
                int j = i + stride;
                bool desc = ((i & size) == 0);
                float a = s[i], c = s[j];
                int   ia = si[i], ic = si[j];
                bool iAfterJ = (a < c) || (a == c && ia > ic);
                bool doswap = desc ? iAfterJ : !iAfterJ;
                if (doswap) { s[i] = c; s[j] = a; si[i] = ic; si[j] = ia; }
            }
            __syncthreads();
        }
    }
    for (int k = tid; k < KSEL; k += 512) {
        oscore[(size_t)b * KSEL + k] = s[k];
        oidxf [(size_t)b * KSEL + k] = (float)si[k];
        oidxi [(size_t)b * KSEL + k] = si[k];
    }
}

// ---------------------------------------------------------------------------
// Projection: compressed[b,k,:] = gather(V)[32 rows] @ W_proj + b_proj
// 2M x 2N register-tiled WMMA as in score_kernel.
// ---------------------------------------------------------------------------
__global__ __launch_bounds__(32) void proj_kernel(
    const float* __restrict__ V, const int* __restrict__ sel,
    const _Float16* __restrict__ WprojT, const float* __restrict__ bproj,
    float* __restrict__ out)
{
    __shared__ _Float16 tile[32 * LDS_STRIDE];
    __shared__ int rows[32];

    const int t    = blockIdx.x;
    const int b    = t >> 6;            // K/32 = 64 super-tiles per batch
    const int mt   = t & 63;
    const int lane = threadIdx.x;
    const int lm   = lane & 15;
    const int half = lane >> 4;

    rows[lane] = sel[(size_t)b * KSEL + mt * 32 + lane];
    __syncthreads();

    for (int m = 0; m < 32; ++m) {
        const float* src = V + ((size_t)b * NN + rows[m]) * DD;
        #pragma unroll
        for (int j = 0; j < 24; ++j)
            tile[m * LDS_STRIDE + lane + 32 * j] = (_Float16)src[lane + 32 * j];
    }
    __syncthreads();

    const _Float16* tbase0 = &tile[lm * LDS_STRIDE + half * 16];
    const _Float16* tbase1 = tbase0 + 16 * LDS_STRIDE;

    for (int nt2 = 0; nt2 < 24; ++nt2) {
        v8f a00 = {}, a01 = {}, a10 = {}, a11 = {};
        const _Float16* bp0 = WprojT + (size_t)(nt2 * 32 + lm) * DD + half * 16;
        const _Float16* bp1 = bp0 + (size_t)16 * DD;
        __builtin_prefetch(bp0 + (size_t)32 * DD, 0, 0);
        #pragma unroll 2
        for (int ks = 0; ks < 24; ++ks) {
            v16h A0 = load_afrag(tbase0, ks);
            v16h A1 = load_afrag(tbase1, ks);
            v16h B0 = *(const v16h*)(bp0 + ks * 32);
            v16h B1 = *(const v16h*)(bp1 + ks * 32);
            a00 = wmma16(A0, B0, a00);
            a01 = wmma16(A0, B1, a01);
            a10 = wmma16(A1, B0, a10);
            a11 = wmma16(A1, B1, a11);
        }
        int n0 = nt2 * 32 + lm, n1 = n0 + 16;
        float bn0 = bproj[n0], bn1 = bproj[n1];
        size_t base = ((size_t)b * KSEL + mt * 32) * DD;
        #pragma unroll
        for (int r = 0; r < 8; ++r) {
            int m = r + 8 * half;
            out[base + (size_t)m * DD + n0]        = a00[r] + bn0;
            out[base + (size_t)m * DD + n1]        = a01[r] + bn1;
            out[base + (size_t)(16 + m) * DD + n0] = a10[r] + bn0;
            out[base + (size_t)(16 + m) * DD + n1] = a11[r] + bn1;
        }
    }
}

// ---------------------------------------------------------------------------
extern "C" void kernel_launch(void* const* d_in, const int* in_sizes, int n_in,
                              void* d_out, int out_size, void* d_ws, size_t ws_size,
                              hipStream_t stream)
{
    (void)in_sizes; (void)n_in; (void)out_size; (void)ws_size;

    const float* V     = (const float*)d_in[0];
    const float* q     = (const float*)d_in[1];
    const float* Wrel1 = (const float*)d_in[2];
    const float* brel1 = (const float*)d_in[3];
    const float* Wrel2 = (const float*)d_in[4];
    const float* brel2 = (const float*)d_in[5];
    const float* Wimp1 = (const float*)d_in[6];
    const float* bimp1 = (const float*)d_in[7];
    const float* Wimp2 = (const float*)d_in[8];
    const float* bimp2 = (const float*)d_in[9];
    const float* Wproj = (const float*)d_in[10];
    const float* bproj = (const float*)d_in[11];

    // workspace carve-out (~3.1 MB)
    char* ws = (char*)d_ws;
    size_t off = 0;
    auto take = [&](size_t bytes) -> char* {
        char* p = ws + off;
        off += (bytes + 255) & ~(size_t)255;
        return p;
    };
    _Float16* W1aT   = (_Float16*)take((size_t)DD * DD * sizeof(_Float16));
    _Float16* WimpT  = (_Float16*)take((size_t)(DD / 2) * DD * sizeof(_Float16));
    _Float16* WprojT = (_Float16*)take((size_t)DD * DD * sizeof(_Float16));
    float*    qrel   = (float*)   take((size_t)BB * DD * sizeof(float));
    int*      selidx = (int*)     take((size_t)BB * KSEL * sizeof(int));

    // output layout: compressed | selected_indices | selected_scores | final_scores
    float* out    = (float*)d_out;
    float* o_comp = out;
    float* o_idx  = out + (size_t)BB * KSEL * DD;
    float* o_scr  = o_idx + (size_t)BB * KSEL;
    float* o_fs   = o_scr + (size_t)BB * KSEL;

    // 1) pack weights (transposed f16, L2-resident afterwards)
    pack_transpose_f16<<<(DD * DD + 255) / 256, 256, 0, stream>>>(Wrel1, W1aT, DD);
    pack_transpose_f16<<<((DD / 2) * DD + 255) / 256, 256, 0, stream>>>(Wimp1, WimpT, DD / 2);
    pack_transpose_f16<<<(DD * DD + 255) / 256, 256, 0, stream>>>(Wproj, WprojT, DD);

    // 2) question-side relevance term (includes b_rel1)
    qrel_kernel<<<(BB * DD + 255) / 256, 256, 0, stream>>>(q, Wrel1, brel1, qrel);

    // 3) fused scores (WMMA, 2Mx2N tiled)
    score_kernel<<<BB * (NN / 32), 32, 0, stream>>>(
        V, q, W1aT, qrel, Wrel2, brel2, WimpT, bimp1, Wimp2, bimp2, o_fs);

    // 4) per-batch top-k (full bitonic sort, jax tie-break)
    topk_kernel<<<BB, 512, 0, stream>>>(o_fs, o_scr, o_idx, selidx);

    // 5) gather + projection (WMMA, 2Mx2N tiled)
    proj_kernel<<<BB * (KSEL / 32), 32, 0, stream>>>(V, selidx, WprojT, bproj, o_comp);
}
